// RNNModel_19799799234773
// MI455X (gfx1250) — compile-verified, matrix operand order checked by reading, not run on hardware
//
#include <hip/hip_runtime.h>
#include <stdint.h>

// LSTM anomaly detector, persistent-kernel formulation for MI455X (gfx1250).
// B=256, T=1024, I=8, H=164, 4H=656. Sequential over T; parallel over batch
// and gate columns. Per-step GEMM (M=256, N=656, K=224 padded) done with
// v_wmma_f32_16x16x32_f16; weights live in registers (pre-packed fragments),
// state lives in L2. Two agent-scope grid barriers per step. All 28 A-frag
// loads are forced ahead of the WMMA block with sched_barrier so they pipeline
// behind a single L2 latency (staged s_wait_loadcnt instead of 0x0 per WMMA).

typedef __attribute__((ext_vector_type(16))) _Float16 v16h;
typedef __attribute__((ext_vector_type(8)))  float    v8f;

#define B_   256
#define T_   1024
#define I_   8
#define H_   164
#define G4   656          // 4*H
#define NT   41           // 656 / 16 n-tiles
#define KC   7            // 224 / 32 k-chunks (32 input-part + 192 hidden-part)
#define KTOT 224
#define NBLK 41
#define WAVES 8           // 256 threads / wave32

// ---- device-wide barrier: monotonic counter, reset each launch by prep ----
__device__ __forceinline__ void gbar(unsigned* cnt, unsigned& epoch) {
  __syncthreads();
  if (threadIdx.x == 0) {
    ++epoch;
    __threadfence();
    __hip_atomic_fetch_add(cnt, 1u, __ATOMIC_RELEASE, __HIP_MEMORY_SCOPE_AGENT);
    const unsigned target = epoch * (unsigned)NBLK;
    while (__hip_atomic_load(cnt, __ATOMIC_ACQUIRE, __HIP_MEMORY_SCOPE_AGENT) < target) {
      __builtin_amdgcn_s_sleep(1);
    }
    __threadfence();
  }
  __syncthreads();
}

__device__ __forceinline__ float sigm(float z) { return 1.f / (1.f + __expf(-z)); }
// tanh via one v_exp_f32 (f16-level accuracy is plenty for this recurrence)
__device__ __forceinline__ float tanh_fast(float z) {
  float e = __expf(2.f * z);
  return (e - 1.f) / (e + 1.f);
}

// ---------------------------------------------------------------------------
// Prep: pack weights into per-lane-contiguous WMMA B fragments (f16), build
// bias sums, init A-buffer (x0 | h=0), zero c, zero flags[:,0], reset barrier.
// K-pair layout per 16-bit WMMA operand (ISA 7.12.2):
//   half p (v=p/2,q=p%2), hi=lane>=16:  K = 16*(v/4) + 8*hi + 2*(v%4) + q
// ---------------------------------------------------------------------------
__global__ void lstm_prep(const float* __restrict__ x,
                          const float* __restrict__ Wih,
                          const float* __restrict__ Whh,
                          const float* __restrict__ bih,
                          const float* __restrict__ bhh,
                          _Float16* __restrict__ WB,    // [NT*KC][512] halves
                          float* __restrict__ bsum,     // [656]
                          _Float16* __restrict__ hA,    // [256][224]
                          float* __restrict__ c,        // [256][164]
                          float* __restrict__ out,      // preds|flags
                          unsigned* __restrict__ barcnt) {
  int idx = blockIdx.x * blockDim.x + threadIdx.x;
  const int nWB = NT * KC * 512;
  const int nHA = B_ * KTOT;
  const int nC  = B_ * H_;
  if (idx < nWB) {
    int frag = idx >> 9, r = idx & 511;
    int lane = r >> 4, p = r & 15;
    int v = p >> 1, q = p & 1, hi = lane >> 4;
    int nt = frag / KC, kc = frag % KC;
    int K = kc * 32 + (v >> 2) * 16 + hi * 8 + (v & 3) * 2 + q;
    int N = nt * 16 + (lane & 15);
    float val = 0.f;
    if (K < 32) { if (K < I_) val = Wih[N * I_ + K]; }
    else        { int kh = K - 32; if (kh < H_) val = Whh[N * H_ + kh]; }
    WB[idx] = (_Float16)val;
    return;
  }
  idx -= nWB;
  if (idx < nHA) {                       // A buffer: [x0 padded | h=0 padded]
    int row = idx / KTOT, k = idx % KTOT;
    float v = 0.f;
    if (k < I_) v = x[(size_t)row * T_ * I_ + k];
    hA[idx] = (_Float16)v;
    return;
  }
  idx -= nHA;
  if (idx < nC)  { c[idx] = 0.f; return; }
  idx -= nC;
  if (idx < G4)  { bsum[idx] = bih[idx] + bhh[idx]; return; }
  idx -= G4;
  if (idx < B_)  { out[(size_t)B_ * T_ + (size_t)idx * T_] = 0.f; return; } // flags[:,0]=0
  idx -= B_;
  if (idx == 0)  { *barcnt = 0u; }
}

// ---------------------------------------------------------------------------
// Persistent kernel: 41 blocks x 256 threads, all T steps in one launch.
// Block n owns gate columns [16n, 16n+16); each wave computes two m-tiles.
// ---------------------------------------------------------------------------
__global__ __launch_bounds__(256) void lstm_persistent(
    const float* __restrict__ x,
    const _Float16* __restrict__ WB,
    const float* __restrict__ bsum,
    const float* __restrict__ Wout,   // [164]
    const float* __restrict__ bout,   // [1]
    _Float16* __restrict__ hA,        // [256][224]  (x_t | h_{t-1}) as f16
    float* __restrict__ c,            // [256][164]
    float* __restrict__ gates,        // [256][656]
    float* __restrict__ out,          // [256*1024] preds, then [256*1024] flags
    unsigned* __restrict__ barcnt) {
  const int tid  = threadIdx.x;
  const int lane = tid & 31;
  const int wave = tid >> 5;
  const int blk  = blockIdx.x;
  const int l15  = lane & 15;
  const int hi   = lane >> 4;
  unsigned epoch = 0;

  const int   nt    = blk;                       // this block's n-tile
  const float biasv = bsum[nt * 16 + l15];       // per output column
  const float boutv = bout[0];

  const int mt0 = wave;                          // two m-tiles per wave
  const int mt1 = wave + WAVES;
  const _Float16* row0 = hA + (size_t)(mt0 * 16 + l15) * KTOT + hi * 8;
  const _Float16* row1 = hA + (size_t)(mt1 * 16 + l15) * KTOT + hi * 8;

  union Frag { v16h v; uint4 q[2]; };

  // B (weight) fragments: loop-invariant, pinned in registers for all T steps.
  Frag Bf[KC];
  {
    const _Float16* wbase = WB + (size_t)nt * KC * 512 + lane * 16;
    #pragma unroll
    for (int kc = 0; kc < KC; ++kc) {
      Bf[kc].q[0] = *(const uint4*)(wbase + (size_t)kc * 512);
      Bf[kc].q[1] = *(const uint4*)(wbase + (size_t)kc * 512 + 8);
    }
  }

  for (int t = 0; t < T_; ++t) {
    // ---------------- GEMM phase (uniform EXEC, WMMA) ----------------
    v8f acc0, acc1;
    #pragma unroll
    for (int r = 0; r < 8; ++r) { acc0[r] = biasv; acc1[r] = biasv; }

    // Issue ALL 28 A-fragment loads; sched_barrier(0) pins them above the
    // WMMA block so the RA keeps 14 fragments live in distinct registers and
    // the waitcnt pass emits staged s_wait_loadcnt (single exposed latency).
    Frag A0[KC], A1[KC];
    #pragma unroll
    for (int kc = 0; kc < KC; ++kc) {
      A0[kc].q[0] = *(const uint4*)(row0 + kc * 32);
      A0[kc].q[1] = *(const uint4*)(row0 + kc * 32 + 16);
      A1[kc].q[0] = *(const uint4*)(row1 + kc * 32);
      A1[kc].q[1] = *(const uint4*)(row1 + kc * 32 + 16);
    }
    __builtin_amdgcn_sched_barrier(0);   // nothing crosses this point

    #pragma unroll
    for (int kc = 0; kc < KC; ++kc) {
      acc0 = __builtin_amdgcn_wmma_f32_16x16x32_f16(
          false, A0[kc].v, false, Bf[kc].v, (short)0, acc0, false, false);
      acc1 = __builtin_amdgcn_wmma_f32_16x16x32_f16(
          false, A1[kc].v, false, Bf[kc].v, (short)0, acc1, false, false);
    }
    // Grouping hints (VMEM-reads first, then WMMAs) for the region scheduler.
    __builtin_amdgcn_sched_group_barrier(0x020, 28, 0);  // 28 VMEM reads
    __builtin_amdgcn_sched_group_barrier(0x008, 14, 0);  // 14 WMMA

    // D layout: VGPR r -> row M = 16*mt + r + 8*hi, col N = 16*nt + l15
    {
      float* gp0 = gates + (size_t)(mt0 * 16 + hi * 8) * G4 + nt * 16 + l15;
      float* gp1 = gates + (size_t)(mt1 * 16 + hi * 8) * G4 + nt * 16 + l15;
      #pragma unroll
      for (int r = 0; r < 8; ++r) gp0[(size_t)r * G4] = acc0[r];
      #pragma unroll
      for (int r = 0; r < 8; ++r) gp1[(size_t)r * G4] = acc1[r];
    }
    gbar(barcnt, epoch);

    // -------- cell + prediction phase: one wave per batch row --------
    const int gw = blk * WAVES + wave;
    if (gw < B_) {
      const int b = gw;
      const float* gr = gates + (size_t)b * G4;
      float partial = 0.f;
      #pragma unroll
      for (int it = 0; it < 6; ++it) {
        int j = lane + it * 32;
        if (j < H_) {
          float iv = sigm(gr[j]);
          float fv = sigm(gr[H_ + j]);
          float gv = tanh_fast(gr[2 * H_ + j]);
          float ov = sigm(gr[3 * H_ + j]);
          float cc = fv * c[(size_t)b * H_ + j] + iv * gv;
          float hh = ov * tanh_fast(cc);
          c[(size_t)b * H_ + j] = cc;
          hA[(size_t)b * KTOT + 32 + j] = (_Float16)hh;
          partial += hh * Wout[j];
        }
      }
      #pragma unroll
      for (int off = 16; off > 0; off >>= 1)
        partial += __shfl_xor(partial, off, 32);
      const float pred = partial + boutv;          // all lanes hold it
      if (lane == 0) out[(size_t)b * T_ + t] = pred;
      if (t + 1 < T_ && lane < I_) {
        float xv = x[(size_t)b * T_ * I_ + (size_t)(t + 1) * I_ + lane];
        if (lane == 0) {                           // WL == 0
          bool an = fabsf(pred - xv) > 0.1f;
          out[(size_t)B_ * T_ + (size_t)b * T_ + (t + 1)] = an ? 1.f : 0.f;
          hA[(size_t)b * KTOT + 0] = (_Float16)(an ? pred : xv);
        } else {
          hA[(size_t)b * KTOT + lane] = (_Float16)xv;
        }
      }
    }
    gbar(barcnt, epoch);
  }
}

extern "C" void kernel_launch(void* const* d_in, const int* in_sizes, int n_in,
                              void* d_out, int out_size, void* d_ws, size_t ws_size,
                              hipStream_t stream) {
  (void)in_sizes; (void)n_in; (void)out_size; (void)ws_size;
  const float* x    = (const float*)d_in[0];
  const float* Wih  = (const float*)d_in[1];
  const float* Whh  = (const float*)d_in[2];
  const float* bih  = (const float*)d_in[3];
  const float* bhh  = (const float*)d_in[4];
  const float* Wout = (const float*)d_in[5];
  const float* bout = (const float*)d_in[6];
  float* out = (float*)d_out;

  char* ws = (char*)d_ws;
  size_t off = 0;
  auto alloc = [&](size_t bytes) -> void* {
    void* p = ws + off;
    off = (off + bytes + 255) & ~(size_t)255;
    return p;
  };
  unsigned*  barcnt = (unsigned*)alloc(256);
  _Float16*  WB     = (_Float16*)alloc((size_t)NT * KC * 512 * sizeof(_Float16));
  float*     bsum   = (float*)   alloc((size_t)G4 * sizeof(float));
  _Float16*  hA     = (_Float16*)alloc((size_t)B_ * KTOT * sizeof(_Float16));
  float*     c      = (float*)   alloc((size_t)B_ * H_ * sizeof(float));
  float*     gates  = (float*)   alloc((size_t)B_ * G4 * sizeof(float));

  const int prepN = NT * KC * 512 + B_ * KTOT + B_ * H_ + G4 + B_ + 1;
  lstm_prep<<<(prepN + 255) / 256, 256, 0, stream>>>(
      x, Wih, Whh, bih, bhh, WB, bsum, hA, c, out, barcnt);
  lstm_persistent<<<NBLK, 256, 0, stream>>>(
      x, WB, bsum, Wout, bout, hA, c, gates, out, barcnt);
}
